// Self_Dynamic_Prototype_51934744543765
// MI455X (gfx1250) — matrix operand order repaired and checked by reading, not compiled, not run on hardware
//
#include <hip/hip_runtime.h>

#define B_   8
#define C_   256
#define HW_  4096
#define HD_  128
#define M_   10
#define NOUT (B_*C_*HW_)

typedef __attribute__((ext_vector_type(16))) _Float16 v16h;
typedef __attribute__((ext_vector_type(8)))  _Float16 v8h;
typedef __attribute__((ext_vector_type(8)))  float    v8f;

static __device__ __forceinline__ v8f wmma16(v16h a, v16h b, v8f c) {
  return __builtin_amdgcn_wmma_f32_16x16x32_f16(false, a, false, b, (short)0, c, false, false);
}
static __device__ __forceinline__ v16h cat8(v8h lo, v8h hi) {
  return __builtin_shufflevector(lo, hi, 0,1,2,3,4,5,6,7,8,9,10,11,12,13,14,15);
}

// ---------------- small prep kernels ----------------

__global__ void k_init(float* feaAcc) {
  if (threadIdx.x == 0 && blockIdx.x == 0) feaAcc[0] = 0.f;
}

__global__ void k_cvt_weights(const float* Wt, const float* Wp, const float* Wg, const float* Wpc,
                              const float* bt, const float* bp, const float* bg, const float* bpc,
                              const float* Wo, _Float16* WhAll, float* biasAll, _Float16* Woh) {
  int i = blockIdx.x * blockDim.x + threadIdx.x;      // 512*256 threads exactly
  if (i < 512 * 256) {
    int o = i >> 8, c = i & 255;
    const float* src = (o < 128) ? Wt : (o < 256) ? Wp : (o < 384) ? Wg : Wpc;
    WhAll[i] = (_Float16)src[(o & 127) * 256 + c];
  }
  if (i < 512) {
    const float* sb = (i < 128) ? bt : (i < 256) ? bp : (i < 384) ? bg : bpc;
    biasAll[i] = sb[i & 127];
  }
  if (i < 256 * 128) Woh[i] = (_Float16)Wo[i];
}

// key [B,C,HW] f32 -> keyT [B,HW,C] f16
__global__ void k_cvt_keyT(const float* key, _Float16* keyT) {
  long long i = (long long)blockIdx.x * 256 + threadIdx.x;  // B*C*HW
  int n = (int)(i & (HW_ - 1));
  int c = (int)((i >> 12) & 255);
  int b = (int)(i >> 20);
  keyT[(((long long)b * HW_ + n) << 8) + c] = (_Float16)key[i];
}

// ---------------- fused projection GEMM (theta/phi/g/pc) via WMMA ----------------
// out[o,n] = sum_c WhAll[o,c]*keyT[n,c] + bias[o]  (per batch), o in [0,512)
__global__ __launch_bounds__(128) void k_proj(const _Float16* __restrict__ WhAll,
                                              const float* __restrict__ biasAll,
                                              const _Float16* __restrict__ keyT,
                                              _Float16* thetaT, _Float16* phiT,
                                              _Float16* Gm, float* keyf) {
  int wid  = blockIdx.x * 4 + (threadIdx.x >> 5);
  int lane = threadIdx.x & 31;
  int b    = wid >> 13;                // 32 o-tiles * 256 n-tiles per batch
  int rem  = wid & 8191;
  int o0   = (rem >> 8) << 4;
  int n0   = (rem & 255) << 4;
  int col  = lane & 15;
  int kA   = (lane < 16) ? 0 : 8;
  int kB   = (lane < 16) ? 0 : 16;
  const _Float16* arow = WhAll + ((long long)(o0 + col) << 8);
  const _Float16* brow = keyT + (((long long)b * HW_ + n0 + col) << 8);
  v8f acc = {};
  for (int k = 0; k < 256; k += 32) {
    v16h A  = *(const v16h*)(arow + k + kB);
    v16h Bt = *(const v16h*)(brow + k + kB);
    acc = wmma16(A, Bt, acc);
  }
  int rb = (lane < 16) ? 0 : 8;
  for (int r = 0; r < 8; ++r) {
    int o = o0 + r + rb;
    int n = n0 + col;
    float v = acc[r] + biasAll[o];
    if (o < 128)      thetaT[(((long long)b * HW_ + n) << 7) + o] = (_Float16)v;
    else if (o < 256) phiT[(((long long)b * HW_ + n) << 7) + (o - 128)] = (_Float16)v;
    else if (o < 384) Gm[(((long long)b * HD_ + (o - 256)) << 12) + n] = (_Float16)v;
    else              keyf[(((long long)b * HW_ + n) << 7) + (o - 384)] = v;
  }
}

// ---------------- BatchNorm (train) over [B*HW, HD] last-dim channels ----------------
__global__ void k_bn_stats_last(const float* x, float* mean, float* inv) {
  int c = blockIdx.x;
  float s = 0.f, s2 = 0.f;
  for (int i = threadIdx.x; i < B_ * HW_; i += 256) {
    float v = x[((long long)i << 7) + c];
    s += v; s2 += v * v;
  }
  __shared__ float sh1[256], sh2[256];
  sh1[threadIdx.x] = s; sh2[threadIdx.x] = s2; __syncthreads();
  for (int st = 128; st > 0; st >>= 1) {
    if (threadIdx.x < st) { sh1[threadIdx.x] += sh1[threadIdx.x + st]; sh2[threadIdx.x] += sh2[threadIdx.x + st]; }
    __syncthreads();
  }
  if (threadIdx.x == 0) {
    float m = sh1[0] / (float)(B_ * HW_);
    float var = sh2[0] / (float)(B_ * HW_) - m * m;
    mean[c] = m; inv[c] = rsqrtf(var + 1e-5f);
  }
}

__global__ void k_bn_relu_last(float* x, const float* mean, const float* inv,
                               const float* gamma, const float* beta) {
  long long i = (long long)blockIdx.x * 256 + threadIdx.x;  // B*HW*HD
  int c = (int)(i & 127);
  float v = (x[i] - mean[c]) * inv[c] * gamma[c] + beta[c];
  x[i] = v > 0.f ? v : 0.f;
}

// ---------------- fused flash attention: nlsa[d,n] = sum_m softmax_m(theta^T phi)[n,m] g[d,m] ----------------
__global__ __launch_bounds__(128) void k_flash(const _Float16* __restrict__ thetaT,
                                               const _Float16* __restrict__ phiT,
                                               const _Float16* __restrict__ Gm,
                                               float* __restrict__ nlsa) {
  int wid  = blockIdx.x * 4 + (threadIdx.x >> 5);
  int lane = threadIdx.x & 31;
  int b    = wid >> 8;
  int n0   = (wid & 255) << 4;
  int col  = lane & 15;
  int half = (lane < 16) ? 0 : 1;
  int koffA = half ? 8 : 0;
  const _Float16* thT = thetaT + ((long long)b * HW_ << 7);
  const _Float16* phT = phiT + ((long long)b * HW_ << 7);
  const _Float16* G   = Gm + ((long long)b * HD_ << 12);
  // theta B-tiles: column n = n0+col over K(d); lanes 0-15 hold d 0..7,16..23 of chunk, 16-31 hold d 8..15,24..31
  v16h Bth[4];
  for (int k = 0; k < 4; ++k)
    Bth[k] = cat8(*(const v8h*)(thT + ((long long)(n0 + col) << 7) + 32 * k + koffA), *(const v8h*)(thT + ((long long)(n0 + col) << 7) + 32 * k + 16 + koffA));
  v8f zeroV = {};
  v8f O[8];
  for (int c2 = 0; c2 < 8; ++c2) O[c2] = zeroV;
  float rmax = -1e30f, rsum = 0.f;
  for (int m0 = 0; m0 < HW_; m0 += 32) {
    // S'[m,n] = sum_d phi[d,m] theta[d,n]; two 16-row m sub-tiles
    v8f S0 = zeroV, S1 = zeroV;
    const _Float16* ar0 = phT + ((long long)(m0 + col) << 7);
    const _Float16* ar1 = phT + ((long long)(m0 + 16 + col) << 7);
    for (int k = 0; k < 4; ++k) {
      int kb = 32 * k;
      v16h A0 = cat8(*(const v8h*)(ar0 + kb + koffA), *(const v8h*)(ar0 + kb + 16 + koffA));
      S0 = wmma16(A0, Bth[k], S0);
      v16h A1 = cat8(*(const v8h*)(ar1 + kb + koffA), *(const v8h*)(ar1 + kb + 16 + koffA));
      S1 = wmma16(A1, Bth[k], S1);
    }
    // online softmax over m (per column n): per-lane partial + one xor-16 combine
    float tmax = -1e30f;
    for (int r = 0; r < 8; ++r) tmax = fmaxf(tmax, fmaxf(S0[r], S1[r]));
    tmax = fmaxf(tmax, __shfl_xor(tmax, 16, 32));
    float nmax = fmaxf(rmax, tmax);
    float corr = __expf(rmax - nmax);
    rmax = nmax;
    float P0[8], P1[8];
    float ps = 0.f;
    for (int r = 0; r < 8; ++r) {
      P0[r] = __expf(S0[r] - nmax);
      P1[r] = __expf(S1[r] - nmax);
      ps += P0[r] + P1[r];
    }
    ps += __shfl_xor(ps, 16, 32);
    rsum = rsum * corr + ps;
    // repack P (C layout) -> B layout f16: lanes 0-15 get m 0..15, lanes 16-31 get m 16..31
    v16h Bp = {};
    for (int r = 0; r < 8; ++r) {
      float x0 = __shfl_xor(P0[r], 16, 32);
      float x1 = __shfl_xor(P1[r], 16, 32);
      float aV = half ? x1 : P0[r];
      float bV = half ? P1[r] : x0;
      Bp[r] = (_Float16)aV;
      Bp[r + 8] = (_Float16)bV;
    }
    for (int c2 = 0; c2 < 8; ++c2)
      for (int r = 0; r < 8; ++r) O[c2][r] *= corr;
    // O^T[d,n] += g[d, m0..m0+31] * P
    for (int c2 = 0; c2 < 8; ++c2) {
      const _Float16* gr = G + ((long long)(16 * c2 + col) << 12) + m0 + 16 * half;
      v16h Ag = *(const v16h*)gr;
      O[c2] = wmma16(Ag, Bp, O[c2]);
    }
  }
  float invs = 1.0f / rsum;
  for (int c2 = 0; c2 < 8; ++c2)
    for (int r = 0; r < 8; ++r) {
      int d = 16 * c2 + r + 8 * half;
      nlsa[(((long long)b * HD_ + d) << 12) + n0 + col] = O[c2][r] * invs;
    }
}

// ---------------- prototype head path ----------------

__global__ void k_heads(const float* keyf, const float* Wh, float* logits) {
  long long i = (long long)blockIdx.x * 256 + threadIdx.x;  // B*HW*M exactly
  int m = (int)(i % M_);
  long long bn = i / M_;
  const float* kf = keyf + (bn << 7);
  const float* w = Wh + m * HD_;
  float s = 0.f;
  for (int d = 0; d < HD_; ++d) s += kf[d] * w[d];
  logits[i] = s;
}

// softmax over n (axis=1): one block per (b, m); in-place logits -> mhw
__global__ void k_head_softmax(float* lg) {
  int b = blockIdx.x / M_, m = blockIdx.x % M_;
  __shared__ float sh[256];
  float mx = -1e30f;
  for (int n = threadIdx.x; n < HW_; n += 256)
    mx = fmaxf(mx, lg[((long long)(b * HW_ + n)) * M_ + m]);
  sh[threadIdx.x] = mx; __syncthreads();
  for (int st = 128; st > 0; st >>= 1) {
    if (threadIdx.x < st) sh[threadIdx.x] = fmaxf(sh[threadIdx.x], sh[threadIdx.x + st]);
    __syncthreads();
  }
  mx = sh[0]; __syncthreads();
  float s = 0.f;
  for (int n = threadIdx.x; n < HW_; n += 256)
    s += __expf(lg[((long long)(b * HW_ + n)) * M_ + m] - mx);
  sh[threadIdx.x] = s; __syncthreads();
  for (int st = 128; st > 0; st >>= 1) {
    if (threadIdx.x < st) sh[threadIdx.x] += sh[threadIdx.x + st];
    __syncthreads();
  }
  float inv = 1.0f / sh[0];
  for (int n = threadIdx.x; n < HW_; n += 256) {
    long long idx = ((long long)(b * HW_ + n)) * M_ + m;
    lg[idx] = __expf(lg[idx] - mx) * inv;
  }
}

__global__ void k_protos(const float* mhw, const float* keyf, float* protos) {
  int b = blockIdx.x / M_, m = blockIdx.x % M_;
  int d = threadIdx.x;  // 128
  float acc = 0.f;
  for (int n = 0; n < HW_; ++n)
    acc += mhw[((long long)(b * HW_ + n)) * M_ + m] * keyf[(((long long)b * HW_ + n) << 7) + d];
  protos[((b * M_ + m) << 7) + d] = acc;
}

__global__ void k_pn(const float* protos, float* pn) {
  int row = blockIdx.x;  // B*M
  int d = threadIdx.x;   // 128
  float v = protos[(row << 7) + d];
  __shared__ float sh[128];
  sh[d] = v * v; __syncthreads();
  for (int st = 64; st > 0; st >>= 1) {
    if (d < st) sh[d] += sh[d + st];
    __syncthreads();
  }
  float nrm = sqrtf(sh[0]);
  pn[(row << 7) + d] = v / fmaxf(nrm, 1e-12f);
}

// per-n wave: score->softmax->argmax->new_q->l2norm; uqT = new_q + nlsa; fea accumulation
__global__ __launch_bounds__(128) void k_assign(const float* __restrict__ keyf,
                                                const float* __restrict__ pn,
                                                const float* __restrict__ nlsa,
                                                _Float16* __restrict__ uqT, float* feaAcc) {
  int wid  = blockIdx.x * 4 + (threadIdx.x >> 5);
  int lane = threadIdx.x & 31;
  int b = wid >> 12;
  int n = wid & 4095;
  const float* kf = keyf + (((long long)b * HW_ + n) << 7);
  const float* pnb = pn + ((long long)b * M_ << 7);
  float kv[4];
  for (int j = 0; j < 4; ++j) kv[j] = kf[lane + 32 * j];
  float sc[M_];
  for (int m = 0; m < M_; ++m) {
    float p = 0.f;
    for (int j = 0; j < 4; ++j) p += kv[j] * pnb[(m << 7) + lane + 32 * j];
    for (int off = 16; off > 0; off >>= 1) p += __shfl_xor(p, off, 32);
    sc[m] = p;
  }
  float mx = sc[0]; int am = 0;
  for (int m = 1; m < M_; ++m) if (sc[m] > mx) { mx = sc[m]; am = m; }
  float e[M_]; float se = 0.f;
  for (int m = 0; m < M_; ++m) { e[m] = __expf(sc[m] - mx); se += e[m]; }
  float invse = 1.0f / se;
  float nq[4] = {0.f, 0.f, 0.f, 0.f};
  for (int m = 0; m < M_; ++m) {
    float w = e[m] * invse;
    for (int j = 0; j < 4; ++j) nq[j] += w * pnb[(m << 7) + lane + 32 * j];
  }
  float ss = 0.f;
  for (int j = 0; j < 4; ++j) ss += nq[j] * nq[j];
  for (int off = 16; off > 0; off >>= 1) ss += __shfl_xor(ss, off, 32);
  float invn = 1.0f / fmaxf(sqrtf(ss), 1e-12f);
  for (int j = 0; j < 4; ++j) {
    int d = lane + 32 * j;
    float u = nq[j] * invn + nlsa[(((long long)b * HD_ + d) << 12) + n];
    uqT[(((long long)b * HW_ + n) << 7) + d] = (_Float16)u;
  }
  float f = 0.f;
  for (int j = 0; j < 4; ++j) {
    float dd = kv[j] - pnb[(am << 7) + lane + 32 * j];
    f += dd * dd;
  }
  for (int off = 16; off > 0; off >>= 1) f += __shfl_xor(f, off, 32);
  if (lane == 0) atomicAdd(feaAcc, f);
}

// scalar losses -> d_out[N..N+2]  (order: fea, cst, dis)
__global__ void k_scalars(const float* pn, const float* feaAcc, float* outs) {
  __shared__ float sh[256];
  float acc = 0.f;
  for (int i = threadIdx.x; i < B_ * M_ * M_ * HD_; i += 256) {
    int h = i & 127;
    int t = i >> 7;
    int m2 = t % M_; t /= M_;
    int m1 = t % M_;
    int b = t / M_;
    if (h > m2) {
      float d = pn[((b * M_ + m1) << 7) + h] - pn[((b * M_ + m2) << 7) + h];
      float v = 1.0f - d * d;
      if (v > 0.f) acc += v;
    }
  }
  sh[threadIdx.x] = acc; __syncthreads();
  for (int st = 128; st > 0; st >>= 1) {
    if (threadIdx.x < st) sh[threadIdx.x] += sh[threadIdx.x + st];
    __syncthreads();
  }
  if (threadIdx.x == 0)
    outs[2] = sh[0] * (2.0f / (M_ * (M_ - 1))) / (float)(M_ * HD_);
  __syncthreads();
  float acc2 = 0.f;
  for (int i = threadIdx.x; i < (B_ - 1) * M_ * HD_; i += 256) {
    int b = i / (M_ * HD_);
    int r = i % (M_ * HD_);
    float d = pn[(b + 1) * M_ * HD_ + r] - pn[b * M_ * HD_ + r];
    acc2 += d * d;
  }
  sh[threadIdx.x] = acc2; __syncthreads();
  for (int st = 128; st > 0; st >>= 1) {
    if (threadIdx.x < st) sh[threadIdx.x] += sh[threadIdx.x + st];
    __syncthreads();
  }
  if (threadIdx.x == 0) {
    outs[1] = sh[0] / (float)((B_ - 1) * M_);
    outs[0] = feaAcc[0] / (float)((long long)B_ * HW_ * HD_);
  }
}

// ---------------- output conv GEMM via WMMA: pre[o,n] = sum_d Wo[o,d] uqT[n,d] + b_o[o] ----------------
__global__ __launch_bounds__(128) void k_out_gemm(const _Float16* __restrict__ Woh,
                                                  const float* __restrict__ b_o,
                                                  const _Float16* __restrict__ uqT,
                                                  float* __restrict__ pre) {
  int wid  = blockIdx.x * 4 + (threadIdx.x >> 5);
  int lane = threadIdx.x & 31;
  int b    = wid >> 12;               // 16 o-tiles * 256 n-tiles per batch
  int rem  = wid & 4095;
  int o0   = (rem >> 8) << 4;
  int n0   = (rem & 255) << 4;
  int col  = lane & 15;
  int kA   = (lane < 16) ? 0 : 8;
  int kB   = (lane < 16) ? 0 : 16;
  const _Float16* arow = Woh + ((long long)(o0 + col) << 7);
  const _Float16* brow = uqT + (((long long)b * HW_ + n0 + col) << 7);
  v8f acc = {};
  for (int k = 0; k < 128; k += 32) {
    v16h A  = *(const v16h*)(arow + k + kB);
    v16h Bt = *(const v16h*)(brow + k + kB);
    acc = wmma16(A, Bt, acc);
  }
  int rb = (lane < 16) ? 0 : 8;
  for (int r = 0; r < 8; ++r) {
    int o = o0 + r + rb;
    int n = n0 + col;
    pre[(((long long)b * C_ + o) << 12) + n] = acc[r] + b_o[o];
  }
}

__global__ void k_bn_stats_mid(const float* x, float* mean, float* inv) {
  int c = blockIdx.x;  // 256 channels
  float s = 0.f, s2 = 0.f;
  for (int i = threadIdx.x; i < B_ * HW_; i += 256) {
    int b = i >> 12, n = i & 4095;
    float v = x[(((long long)b * C_ + c) << 12) + n];
    s += v; s2 += v * v;
  }
  __shared__ float sh1[256], sh2[256];
  sh1[threadIdx.x] = s; sh2[threadIdx.x] = s2; __syncthreads();
  for (int st = 128; st > 0; st >>= 1) {
    if (threadIdx.x < st) { sh1[threadIdx.x] += sh1[threadIdx.x + st]; sh2[threadIdx.x] += sh2[threadIdx.x + st]; }
    __syncthreads();
  }
  if (threadIdx.x == 0) {
    float m = sh1[0] / (float)(B_ * HW_);
    float var = sh2[0] / (float)(B_ * HW_) - m * m;
    mean[c] = m; inv[c] = rsqrtf(var + 1e-5f);
  }
}

__global__ void k_finalize(float* out, const float* query, const float* mean, const float* inv,
                           const float* gam, const float* bet) {
  long long i = (long long)blockIdx.x * 256 + threadIdx.x;  // NOUT
  int c = (int)((i >> 12) & 255);
  float v = (out[i] - mean[c]) * inv[c] * gam[c] + bet[c];
  v = v > 0.f ? v : 0.f;
  out[i] = v + query[i];
}

// ---------------- host launcher ----------------

extern "C" void kernel_launch(void* const* d_in, const int* in_sizes, int n_in,
                              void* d_out, int out_size, void* d_ws, size_t ws_size,
                              hipStream_t stream) {
  (void)in_sizes; (void)n_in; (void)out_size; (void)ws_size;
  const float* key      = (const float*)d_in[0];
  const float* query    = (const float*)d_in[1];
  const float* W_theta  = (const float*)d_in[2];
  const float* b_theta  = (const float*)d_in[3];
  const float* W_phi    = (const float*)d_in[4];
  const float* b_phi    = (const float*)d_in[5];
  const float* W_g      = (const float*)d_in[6];
  const float* b_g      = (const float*)d_in[7];
  const float* W_pc     = (const float*)d_in[8];
  const float* b_pc     = (const float*)d_in[9];
  const float* gamma_pc = (const float*)d_in[10];
  const float* beta_pc  = (const float*)d_in[11];
  const float* W_heads  = (const float*)d_in[12];
  const float* W_o      = (const float*)d_in[13];
  const float* b_o      = (const float*)d_in[14];
  const float* gamma_o  = (const float*)d_in[15];
  const float* beta_o   = (const float*)d_in[16];

  char* ws = (char*)d_ws;
  size_t off = 0;
  auto take = [&](size_t bytes) -> char* {
    char* p = ws + off;
    off = (off + bytes + 255) & ~(size_t)255;
    return p;
  };
  _Float16* keyT   = (_Float16*)take((size_t)B_ * HW_ * C_ * 2);   // 16 MB
  _Float16* WhAll  = (_Float16*)take(512 * 256 * 2);
  float*    biasAll= (float*)take(512 * 4);
  _Float16* Woh    = (_Float16*)take(256 * 128 * 2);
  _Float16* thetaT = (_Float16*)take((size_t)B_ * HW_ * HD_ * 2);  // 8 MB
  _Float16* phiT   = (_Float16*)take((size_t)B_ * HW_ * HD_ * 2);  // 8 MB
  _Float16* Gm     = (_Float16*)take((size_t)B_ * HD_ * HW_ * 2);  // 8 MB
  float*    keyf   = (float*)take((size_t)B_ * HW_ * HD_ * 4);     // 16 MB
  float*    mean1  = (float*)take(HD_ * 4);
  float*    inv1   = (float*)take(HD_ * 4);
  float*    nlsa   = (float*)take((size_t)B_ * HD_ * HW_ * 4);     // 16 MB
  float*    mhw    = (float*)take((size_t)B_ * HW_ * M_ * 4);      // 1.3 MB
  float*    protos = (float*)take((size_t)B_ * M_ * HD_ * 4);
  float*    pn     = (float*)take((size_t)B_ * M_ * HD_ * 4);
  _Float16* uqT    = (_Float16*)take((size_t)B_ * HW_ * HD_ * 2);  // 8 MB
  float*    mean2  = (float*)take(C_ * 4);
  float*    inv2   = (float*)take(C_ * 4);
  float*    feaAcc = (float*)take(256);

  float* out = (float*)d_out;           // pre-BN conv output accumulated in place
  float* outScalars = out + NOUT;

  k_init<<<1, 32, 0, stream>>>(feaAcc);
  k_cvt_weights<<<512, 256, 0, stream>>>(W_theta, W_phi, W_g, W_pc,
                                         b_theta, b_phi, b_g, b_pc,
                                         W_o, WhAll, biasAll, Woh);
  k_cvt_keyT<<<32768, 256, 0, stream>>>(key, keyT);
  k_proj<<<16384, 128, 0, stream>>>(WhAll, biasAll, keyT, thetaT, phiT, Gm, keyf);
  k_bn_stats_last<<<128, 256, 0, stream>>>(keyf, mean1, inv1);
  k_bn_relu_last<<<16384, 256, 0, stream>>>(keyf, mean1, inv1, gamma_pc, beta_pc);
  k_flash<<<512, 128, 0, stream>>>(thetaT, phiT, Gm, nlsa);
  k_heads<<<1280, 256, 0, stream>>>(keyf, W_heads, mhw);
  k_head_softmax<<<B_ * M_, 256, 0, stream>>>(mhw);
  k_protos<<<B_ * M_, 128, 0, stream>>>(mhw, keyf, protos);
  k_pn<<<B_ * M_, 128, 0, stream>>>(protos, pn);
  k_assign<<<8192, 128, 0, stream>>>(keyf, pn, nlsa, uqT, feaAcc);
  k_scalars<<<1, 256, 0, stream>>>(pn, feaAcc, outScalars);
  k_out_gemm<<<8192, 128, 0, stream>>>(Woh, b_o, uqT, out);
  k_bn_stats_mid<<<256, 256, 0, stream>>>(out, mean2, inv2);
  k_finalize<<<32768, 256, 0, stream>>>(out, query, mean2, inv2, gamma_o, beta_o);
}
